// SSMBlock_83373905150195
// MI455X (gfx1250) — matrix operand order repaired and checked by reading, not compiled, not run on hardware
//
#include <hip/hip_runtime.h>
#include <hip/hip_bf16.h>
#include <cstdint>

// ---------------------------------------------------------------------------
// SSM block for MI455X (gfx1250, wave32, WMMA).
//   B=16, S=4096, IN=N=OUT=256
//   d_out = states[16][4096][256] f32  ++  outputs[16][4096][256] f32
//   d_ws  = BuT[4096][16][256] f32 (time-major input injection, 64 MB)
// ---------------------------------------------------------------------------

#define BDIM 16
#define SDIM 4096
#define NDIM 256
#define LDSPAD 264   // 264 bf16 = 528 B row stride -> bank rotation, 16B aligned
#define TPB 4        // tiles (time steps / row-tiles) per block in k_bu / k_out

typedef __attribute__((ext_vector_type(16))) __bf16 v16bf;
typedef __attribute__((ext_vector_type(8)))  __bf16 v8bf;
typedef __attribute__((ext_vector_type(8)))  float  v8f;

#define WMMA_BF16(a, b, c) \
  __builtin_amdgcn_wmma_f32_16x16x32_bf16(false, (a), false, (b), (short)0, (c), false, false)

#if defined(__gfx1250__) && __has_builtin(__builtin_amdgcn_global_load_async_to_lds_b128) && \
    __has_builtin(__builtin_amdgcn_s_wait_asynccnt)
#define USE_ASYNC 1
#else
#define USE_ASYNC 0
#endif

#if USE_ASYNC
// Builtin signature (from hipcc diagnostic): param0 is 'int __vector_size(16) AS(1)*'.
typedef int v4i_vs __attribute__((__vector_size__(4 * sizeof(int))));
__device__ inline void async_b128(uintptr_t gsrc, uintptr_t ldst) {
  __builtin_amdgcn_global_load_async_to_lds_b128(
      (__attribute__((address_space(1))) v4i_vs*)gsrc,
      (__attribute__((address_space(3))) v4i_vs*)ldst, 0, 0);
}
#endif

// B-operand fragment: 16 contiguous K values of one weight row, f32 -> bf16.
// lane n' = lane&15 selects column; koff = (lane>=16)*16 selects K half.
__device__ inline v16bf load_w_frag_f32(const float* __restrict__ wrow) {
  v16bf r;
#pragma unroll
  for (int i = 0; i < 16; ++i) r[i] = (__bf16)wrow[i];
  return r;
}

// A-operand fragment (16x32 bf16) from an LDS row (stride LDSPAD, 16B aligned).
// lanes 0-15: K = kb..kb+15 ; lanes 16-31: K = kb+16..kb+31 (2*coff).
__device__ inline v16bf lds_a_frag(const __bf16* __restrict__ row, int kb, int coff) {
  const v8bf lo = *(const v8bf*)(row + kb + 2 * coff);
  const v8bf hi = *(const v8bf*)(row + kb + 2 * coff + 8);
  return __builtin_shufflevector(lo, hi, 0, 1, 2, 3, 4, 5, 6, 7, 8, 9, 10, 11, 12, 13, 14, 15);
}

// ---------------------------------------------------------------------------
// Kernel 1: BuT[t][b][n] = sum_i u[b][t][i] * B_w[n][i] + B_b[n]
// grid = S/TPB blocks (TPB time steps each), 512 threads = 16 waves.
// Weight fragments converted to registers once, reused for TPB tiles.
// ---------------------------------------------------------------------------
__global__ __launch_bounds__(512) void k_bu(const float* __restrict__ u,
                                            const float* __restrict__ Bw,
                                            const float* __restrict__ Bb,
                                            float* __restrict__ BuT) {
  __shared__ __bf16 u_bf[TPB][BDIM][LDSPAD];
  const int t0 = blockIdx.x * TPB;
  const int tid = threadIdx.x;
  const int lane = tid & 31;
  const int w = tid >> 5;
  const int nl = lane & 15;
  const int hf = lane >> 4;
  const int coff = hf * 8, koff = hf * 16, mrow0 = hf * 8;
  const int ncol = w * 16 + nl;

  // stage u[:, t0..t0+TPB, :] as bf16 (wave w = row b=w)
#pragma unroll
  for (int tt = 0; tt < TPB; ++tt) {
    const float* src = u + ((size_t)w * SDIM + t0 + tt) * NDIM + lane * 8;
    v8bf tmp;
#pragma unroll
    for (int j = 0; j < 8; ++j) tmp[j] = (__bf16)src[j];
    *(v8bf*)&u_bf[tt][w][lane * 8] = tmp;
  }

  // loop-invariant weight fragments (rows [16w,16w+16) of B_w)
  v16bf wfrag[8];
#pragma unroll
  for (int kb8 = 0; kb8 < 8; ++kb8)
    wfrag[kb8] = load_w_frag_f32(Bw + (size_t)ncol * NDIM + kb8 * 32 + koff);
  const float bias = Bb[ncol];
  __syncthreads();

#pragma unroll
  for (int tt = 0; tt < TPB; ++tt) {
    v16bf afrag[8];
#pragma unroll
    for (int kb8 = 0; kb8 < 8; ++kb8)
      afrag[kb8] = lds_a_frag(u_bf[tt][nl], kb8 * 32, coff);
    v8f acc0, acc1;
#pragma unroll
    for (int r = 0; r < 8; ++r) { acc0[r] = bias; acc1[r] = 0.0f; }
#pragma unroll
    for (int kb8 = 0; kb8 < 8; kb8 += 2) {
      acc0 = WMMA_BF16(afrag[kb8], wfrag[kb8], acc0);
      acc1 = WMMA_BF16(afrag[kb8 + 1], wfrag[kb8 + 1], acc1);
    }
    float* dst = BuT + (size_t)(t0 + tt) * (BDIM * NDIM);
#pragma unroll
    for (int r = 0; r < 8; ++r) dst[(mrow0 + r) * NDIM + ncol] = acc0[r] + acc1[r];
  }
}

// ---------------------------------------------------------------------------
// Kernel 2: sequential scan. grid=1 block, 512 threads = 16 waves.
//   s_t = LN(s_{t-1} @ A^T + Bu_t) -> relu -> states[b][t][n]
// Wave w holds its 16 rows of A in registers (8 x v16bf) for all 4096 steps.
// Bu_t tiles are prefetched one step ahead via async-to-LDS (double buffered).
// ---------------------------------------------------------------------------
__global__ __launch_bounds__(512) void k_scan(const float* __restrict__ BuT,
                                              const float* __restrict__ A,
                                              const float* __restrict__ ln_g,
                                              const float* __restrict__ ln_b,
                                              float* __restrict__ states) {
  __shared__ __bf16 s_bf[BDIM][LDSPAD];   // state, bf16, for next step's A-operand
  __shared__ float stg[16][16][16];       // per-wave D tiles for LN reduction
#if USE_ASYNC
  __shared__ float bu_buf[2][BDIM][NDIM]; // double-buffered Bu_t tiles (16 KB each)
#endif
  const int tid = threadIdx.x;
  const int lane = tid & 31;
  const int w = tid >> 5;
  const int nl = lane & 15;
  const int hf = lane >> 4;
  const int coff = hf * 8, koff = hf * 16, mrow0 = hf * 8;
  const int ncol = w * 16 + nl;

  // Loop-invariant B-operand: rows [16w,16w+16) of A, bf16, in 64 VGPRs.
  v16bf bfrag[8];
#pragma unroll
  for (int kb8 = 0; kb8 < 8; ++kb8)
    bfrag[kb8] = load_w_frag_f32(A + (size_t)ncol * NDIM + kb8 * 32 + koff);

  // zero initial state
  for (int i = tid; i < BDIM * LDSPAD; i += 512) (&s_bf[0][0])[i] = (__bf16)0.0f;

  // LN params for this thread's columns (wave w handles row b=w; lane -> n=lane+32j)
  float g[8], bb[8];
#pragma unroll
  for (int j = 0; j < 8; ++j) {
    g[j] = ln_g[lane + 32 * j];
    bb[j] = ln_b[lane + 32 * j];
  }

#if USE_ASYNC
  {  // prefetch t=0 into buffer 0: 2 x b128 per thread covers the 16 KB tile
    const uintptr_t gsrc = (uintptr_t)(BuT) + (size_t)tid * 16;
    const uintptr_t ldst = (uintptr_t)(&bu_buf[0][0][0]) + (size_t)tid * 16;
    async_b128(gsrc, ldst);
    async_b128(gsrc + 8192, ldst + 8192);
  }
#else
  v8f bu_next;
  {
    const float* p = BuT + ncol;
#pragma unroll
    for (int r = 0; r < 8; ++r) bu_next[r] = p[(mrow0 + r) * NDIM];
  }
#endif
  __syncthreads();

  for (int t = 0; t < SDIM; ++t) {
    const int cur = t & 1;
    v8f acc0, acc1;
#if USE_ASYNC
    if (t + 1 < SDIM) {  // prefetch next tile into the other buffer
      const uintptr_t gsrc =
          (uintptr_t)(BuT + (size_t)(t + 1) * (BDIM * NDIM)) + (size_t)tid * 16;
      const uintptr_t ldst = (uintptr_t)(&bu_buf[cur ^ 1][0][0]) + (size_t)tid * 16;
      async_b128(gsrc, ldst);
      async_b128(gsrc + 8192, ldst + 8192);
      __builtin_amdgcn_s_wait_asynccnt(2);  // oldest two (tile t) done
    } else {
      __builtin_amdgcn_s_wait_asynccnt(0);
    }
    __syncthreads();  // publish bu_buf[cur] to all waves
#pragma unroll
    for (int r = 0; r < 8; ++r) acc0[r] = bu_buf[cur][mrow0 + r][ncol];
#else
    acc0 = bu_next;
    if (t + 1 < SDIM) {
      const float* p = BuT + (size_t)(t + 1) * (BDIM * NDIM) + ncol;
#pragma unroll
      for (int r = 0; r < 8; ++r) bu_next[r] = p[(mrow0 + r) * NDIM];
    }
#endif
    // Preload all A-fragments (state) so no LDS waits sit inside the WMMA chain.
    v16bf afrag[8];
#pragma unroll
    for (int kb8 = 0; kb8 < 8; ++kb8)
      afrag[kb8] = lds_a_frag(s_bf[nl], kb8 * 32, coff);
#pragma unroll
    for (int r = 0; r < 8; ++r) acc1[r] = 0.0f;

    // D = s_bf(16x256) x A^T-tile + Bu : two independent 4-deep WMMA chains
#pragma unroll
    for (int kb8 = 0; kb8 < 8; kb8 += 2) {
      acc0 = WMMA_BF16(afrag[kb8], bfrag[kb8], acc0);
      acc1 = WMMA_BF16(afrag[kb8 + 1], bfrag[kb8 + 1], acc1);
    }

    // stage D tile for cross-wave LayerNorm reduction
#pragma unroll
    for (int r = 0; r < 8; ++r) stg[w][mrow0 + r][nl] = acc0[r] + acc1[r];
    __syncthreads();

    // wave w <-> batch row b=w; lane covers n = lane + 32j
    float x[8], sum = 0.f, sq = 0.f;
#pragma unroll
    for (int j = 0; j < 8; ++j) {
      const int n = lane + 32 * j;
      const float v = stg[n >> 4][w][n & 15];
      x[j] = v;
      sum += v;
      sq += v * v;
    }
#pragma unroll
    for (int o = 16; o >= 1; o >>= 1) {
      sum += __shfl_xor(sum, o, 32);
      sq += __shfl_xor(sq, o, 32);
    }
    const float mean = sum * (1.0f / 256.0f);
    const float var = sq * (1.0f / 256.0f) - mean * mean;
    const float rstd = rsqrtf(var + 1e-5f);

    float* srow = states + ((size_t)w * SDIM + t) * NDIM;
#pragma unroll
    for (int j = 0; j < 8; ++j) {
      const int n = lane + 32 * j;
      float y = (x[j] - mean) * rstd * g[j] + bb[j];
      y = fmaxf(y, 0.0f);
      srow[n] = y;                 // f32 states output
      s_bf[w][n] = (__bf16)y;      // bf16 state for next step
    }
    __syncthreads();
  }
}

// ---------------------------------------------------------------------------
// Kernel 3: outputs[r][o] = sum_n states[r][n] * C_w[o][n] + C_b[o]
// grid = 65536/(16*TPB) blocks, 512 threads = 16 waves (one o-tile each).
// Weight fragments converted to registers once, reused for TPB row-tiles.
// ---------------------------------------------------------------------------
__global__ __launch_bounds__(512) void k_out(const float* __restrict__ states,
                                             const float* __restrict__ Cw,
                                             const float* __restrict__ Cb,
                                             float* __restrict__ out) {
  __shared__ __bf16 s_act[TPB][BDIM][LDSPAD];
  const int tid = threadIdx.x;
  const int lane = tid & 31;
  const int w = tid >> 5;
  const size_t row0 = (size_t)blockIdx.x * (16 * TPB);
  const int nl = lane & 15;
  const int hf = lane >> 4;
  const int coff = hf * 8, koff = hf * 16, mrow0 = hf * 8;
  const int ocol = w * 16 + nl;

#pragma unroll
  for (int tt = 0; tt < TPB; ++tt) {
    const float* src = states + (row0 + tt * 16 + w) * NDIM + lane * 8;
    v8bf tmp;
#pragma unroll
    for (int j = 0; j < 8; ++j) tmp[j] = (__bf16)src[j];
    *(v8bf*)&s_act[tt][w][lane * 8] = tmp;
  }

  v16bf wfrag[8];
#pragma unroll
  for (int kb8 = 0; kb8 < 8; ++kb8)
    wfrag[kb8] = load_w_frag_f32(Cw + (size_t)ocol * NDIM + kb8 * 32 + koff);
  const float bias = Cb[ocol];
  __syncthreads();

#pragma unroll
  for (int tt = 0; tt < TPB; ++tt) {
    v16bf afrag[8];
#pragma unroll
    for (int kb8 = 0; kb8 < 8; ++kb8)
      afrag[kb8] = lds_a_frag(s_act[tt][nl], kb8 * 32, coff);
    v8f acc0, acc1;
#pragma unroll
    for (int r = 0; r < 8; ++r) { acc0[r] = bias; acc1[r] = 0.0f; }
#pragma unroll
    for (int kb8 = 0; kb8 < 8; kb8 += 2) {
      acc0 = WMMA_BF16(afrag[kb8], wfrag[kb8], acc0);
      acc1 = WMMA_BF16(afrag[kb8 + 1], wfrag[kb8 + 1], acc1);
    }
    float* dst = out + (row0 + tt * 16) * NDIM;
#pragma unroll
    for (int r = 0; r < 8; ++r) dst[(mrow0 + r) * NDIM + ocol] = acc0[r] + acc1[r];
  }
}

// ---------------------------------------------------------------------------
extern "C" void kernel_launch(void* const* d_in, const int* in_sizes, int n_in,
                              void* d_out, int out_size, void* d_ws, size_t ws_size,
                              hipStream_t stream) {
  (void)in_sizes; (void)n_in; (void)out_size; (void)ws_size;
  const float* u    = (const float*)d_in[0];
  const float* A    = (const float*)d_in[1];
  const float* Bw   = (const float*)d_in[2];
  const float* Bb   = (const float*)d_in[3];
  const float* ln_g = (const float*)d_in[4];
  const float* ln_b = (const float*)d_in[5];
  const float* Cw   = (const float*)d_in[6];
  const float* Cb   = (const float*)d_in[7];

  float* states  = (float*)d_out;                                   // [16][4096][256]
  float* outputs = (float*)d_out + (size_t)BDIM * SDIM * NDIM;      // [16][4096][256]
  float* BuT     = (float*)d_ws;                                    // [4096][16][256]

  k_bu<<<SDIM / TPB, 512, 0, stream>>>(u, Bw, Bb, BuT);
  k_scan<<<1, 512, 0, stream>>>(BuT, A, ln_g, ln_b, states);
  k_out<<<(BDIM * SDIM) / (16 * TPB), 512, 0, stream>>>(states, Cw, Cb, outputs);
}
